// ViTSelfAttentionOperator_19000935317648
// MI455X (gfx1250) — compile-verified, matrix operand order checked by reading, not run on hardware
//
#include <hip/hip_runtime.h>
#include <hip/hip_bf16.h>
#include <math.h>

// ---------------------------------------------------------------------------
// Frequency-domain FNO attention for gfx1250 (wave32, WMMA bf16 16x16x32).
// All GEMM operands laid out so every fragment load is 2x16B contiguous.
// ---------------------------------------------------------------------------

typedef __attribute__((ext_vector_type(16))) __bf16 v16bf;
typedef __attribute__((ext_vector_type(8)))  float  v8f;
typedef __attribute__((ext_vector_type(4)))  int    si4;
typedef __attribute__((ext_vector_type(8)))  int    si8;

static constexpr int B_   = 2;
static constexpr int S_   = 64;
static constexpr int D_   = 64;     // per-head dim
static constexpr int NH_  = 8;
static constexpr int HD_  = 512;    // NH*D
static constexpr int M2_  = 256;    // 16x16 retained modes
static constexpr int BS_  = 128;    // B*S
static constexpr int BH_  = 16;     // B*NH
static constexpr int FEAT_ = D_ * M2_ * 2;  // 32768 (d, mode, re/im)

#define PI_F 3.14159265358979323846f

// ----------------------------- WMMA helpers --------------------------------

__device__ inline v8f wmma_bf16(v16bf a, v16bf b, v8f c) {
  return __builtin_amdgcn_wmma_f32_16x16x32_bf16(
      false, a, false, b, (short)0, c, false, false);
}

// K index inside a 16-element bf16 fragment (ISA 16-bit operand layout).
// Per lane the 16 elements form two contiguous 8-element (16B) runs.
__device__ inline int frag_k(int e, int half) {
  int j = e >> 1, p = e & 1;
  return ((j & 3) * 2 + p) + 8 * half + ((j >> 2) << 4);
}

// A tile: 16x32 (MxK), src row-major [M][K], leading dim ld.
__device__ inline v16bf load_a16x32(const __bf16* __restrict__ src, int ld) {
  int lane = threadIdx.x & 31;
  int row = lane & 15, half = lane >> 4;
  v16bf a;
#pragma unroll
  for (int e = 0; e < 16; ++e) a[e] = src[row * ld + frag_k(e, half)];
  return a;
}

// B tile: 32x16 (KxN) taken from a [N][K] row-major (transposed) source.
__device__ inline v16bf load_bT16(const __bf16* __restrict__ src, int ld) {
  int lane = threadIdx.x & 31;
  int col = lane & 15, half = lane >> 4;
  v16bf b;
#pragma unroll
  for (int e = 0; e < 16; ++e) b[e] = src[col * ld + frag_k(e, half)];
  return b;
}

// --------------------- Tensor Data Mover (TDM) helper ----------------------
// 2D tile load global -> LDS. D# per CDNA5 ISA.
__device__ inline void tdm_load_2d_to_lds(unsigned lds_off, const void* gptr,
                                          unsigned tensor_d0, unsigned tensor_d1,
                                          unsigned tile_d0, unsigned tile_d1,
                                          unsigned long long stride0) {
  unsigned long long ga = (unsigned long long)gptr;
  si4 g0;
  g0[0] = 1;                                   // count=1, user descriptor
  g0[1] = (int)lds_off;                        // lds_addr (bytes)
  g0[2] = (int)(ga & 0xffffffffull);           // global addr [95:64]
  g0[3] = (int)((ga >> 32) & 0x01ffffffull) | (int)(2u << 30);  // + type=2
  si8 g1;
  g1[0] = (int)(1u << 16);                     // wg_mask=0, data_size=1 (2B)
  g1[1] = (int)((tensor_d0 & 0xffffu) << 16);  // barrier=0 | dim0 lo16
  g1[2] = (int)((tensor_d0 >> 16) | ((tensor_d1 & 0xffffu) << 16));
  g1[3] = (int)((tensor_d1 >> 16) | (tile_d0 << 16));
  g1[4] = (int)(tile_d1 & 0xffffu);            // tile_dim1 | tile_dim2=0
  g1[5] = (int)(stride0 & 0xffffffffull);      // dim0 stride lo32
  g1[6] = (int)((stride0 >> 32) & 0xffffull);  // dim0 stride hi16 | d1s lo=0
  g1[7] = 0;
  asm volatile("tensor_load_to_lds %0, %1" : : "s"(g0), "s"(g1) : "memory");
}

// ------------------------ 1. weight conversion -----------------------------
// in : float [Ci][Co][16][16][2]
// out: bf16  [m][r][Co][Ci]   (transposed per-mode B operand)
__global__ void k_cvt_w(const float* __restrict__ w, __bf16* __restrict__ out,
                        int Ci, int Co) {
  long long tid = (long long)blockIdx.x * blockDim.x + threadIdx.x;
  long long total = (long long)Ci * Co * M2_ * 2;
  if (tid >= total) return;
  long long t = tid;
  int r = (int)(t & 1); t >>= 1;
  int m = (int)(t & 255); t >>= 8;
  int co = (int)(t % Co);
  int ci = (int)(t / Co);
  out[(((long long)(m * 2 + r) * Co + co) * Ci + ci)] = (__bf16)w[tid];
}

// ------------------------ 2. forward truncated DFT -------------------------
// seq plane (32x32) -> 16x16 complex modes; out layout bf16 [m][r][bs][ci]
__global__ void k_dft_fwd(const float* __restrict__ seq,
                          __bf16* __restrict__ xfb) {
  int plane = blockIdx.x;           // bs*64 + ci
  int bs = plane >> 6, ci = plane & 63;
  __shared__ float p[1024];
  __shared__ float tr[16 * 32], ti[16 * 32];
  const float* src = seq + (long long)plane * 1024;
  for (int i = threadIdx.x; i < 1024; i += 256) p[i] = src[i];
  __syncthreads();
  for (int i = threadIdx.x; i < 512; i += 256) {
    int kx = i >> 5, y = i & 31;
    float ar = 0.f, ai = 0.f;
    for (int x = 0; x < 32; ++x) {
      float ang = -2.f * PI_F * (float)(kx * x) / 32.f;
      float s, c; __sincosf(ang, &s, &c);
      float v = p[x * 32 + y];
      ar += v * c; ai += v * s;
    }
    tr[i] = ar; ti[i] = ai;
  }
  __syncthreads();
  {
    int i = threadIdx.x;
    int kx = i >> 4, ky = i & 15;
    float ar = 0.f, ai = 0.f;
    for (int y = 0; y < 32; ++y) {
      float ang = -2.f * PI_F * (float)(ky * y) / 32.f;
      float s, c; __sincosf(ang, &s, &c);
      float vr = tr[kx * 32 + y], vi = ti[kx * 32 + y];
      ar += vr * c - vi * s;
      ai += vr * s + vi * c;
    }
    int m = kx * 16 + ky;
    xfb[((long long)(m * 2 + 0) * BS_ + bs) * 64 + ci] = (__bf16)ar;
    xfb[((long long)(m * 2 + 1) * BS_ + bs) * 64 + ci] = (__bf16)ai;
  }
}

// ------------------ 3. per-mode complex channel mix (Q/K/V) ----------------
// One wave: 16 rows x 128 cols (8 N-tiles) per mode; A fragments hoisted.
__global__ void k_gemm_modes(const __bf16* __restrict__ xfb,
                             const __bf16* __restrict__ wqb,
                             const __bf16* __restrict__ wkb,
                             const __bf16* __restrict__ wvb,
                             __bf16* __restrict__ qfb,
                             __bf16* __restrict__ kfb,
                             __bf16* __restrict__ vfb) {
  int z = blockIdx.z;
  int which = z % 3;
  int m = z / 3;
  int slab = blockIdx.x;            // 0..3 : 128-wide co slab
  int mt = blockIdx.y;              // 0..7 : 16-row bs tile
  const __bf16* wb = (which == 0) ? wqb : (which == 1 ? wkb : wvb);

  const __bf16* Ar = xfb + ((long long)(m * 2 + 0) * BS_ + mt * 16) * 64;
  const __bf16* Ai = xfb + ((long long)(m * 2 + 1) * BS_ + mt * 16) * 64;
  v16bf ar0 = load_a16x32(Ar, 64),      ar1 = load_a16x32(Ar + 32, 64);
  v16bf ai0 = load_a16x32(Ai, 64),      ai1 = load_a16x32(Ai + 32, 64);

  float oscale = 1.0f;
  if (which == 0) {
    int ky = m & 15;                  // Parseval weight + attn scale
    oscale = ((ky == 0) ? 1.0f : 2.0f) * (1.0f / 1024.0f) * (1.0f / 256.0f);
  }
  int lane = threadIdx.x & 31;
  int col = lane & 15, rb = (lane >> 4) * 8;

#pragma unroll
  for (int nt = 0; nt < 8; ++nt) {
    int co0 = slab * 128 + nt * 16;
    const __bf16* BrT = wb + ((long long)(m * 2 + 0) * HD_ + co0) * 64;
    const __bf16* BiT = wb + ((long long)(m * 2 + 1) * HD_ + co0) * 64;
    v16bf br0 = load_bT16(BrT, 64),   br1 = load_bT16(BrT + 32, 64);
    v16bf bi0 = load_bT16(BiT, 64),   bi1 = load_bT16(BiT + 32, 64);
    v8f rr = {}, ii = {}, im = {};
    rr = wmma_bf16(ar0, br0, rr);  rr = wmma_bf16(ar1, br1, rr);
    ii = wmma_bf16(ai0, bi0, ii);  ii = wmma_bf16(ai1, bi1, ii);
    im = wmma_bf16(ar0, bi0, im);  im = wmma_bf16(ar1, bi1, im);
    im = wmma_bf16(ai0, br0, im);  im = wmma_bf16(ai1, br1, im);
#pragma unroll
    for (int g = 0; g < 8; ++g) {
      int bs = mt * 16 + rb + g;
      int o = co0 + col;
      int b = bs >> 6, s = bs & 63;
      int h = o >> 6, dd = o & 63;
      float re = (rr[g] - ii[g]) * oscale;
      float imv = im[g] * oscale;
      if (which == 2) {
        // vfb transposed: [bh][feature][s]
        long long base =
            ((long long)(b * NH_ + h) * FEAT_ + (dd * (M2_ * 2) + m * 2)) * 64 + s;
        vfb[base] = (__bf16)re;
        vfb[base + 64] = (__bf16)imv;
      } else {
        __bf16* outp = (which == 0) ? qfb : kfb;
        long long base =
            ((((long long)(b * NH_ + h) * S_ + s) * D_ + dd) * M2_ + m) * 2;
        outp[base + 0] = (__bf16)re;
        outp[base + 1] = (__bf16)imv;
      }
    }
  }
}

// ------------------------- 4. log-CPB bias MLP -----------------------------
__global__ void k_bias(const float* __restrict__ w1, const float* __restrict__ b1,
                       const float* __restrict__ w2, const int* __restrict__ nsx,
                       const int* __restrict__ nsy, float* __restrict__ bias) {
  int t = blockIdx.x * blockDim.x + threadIdx.x;
  if (t >= S_ * S_) return;
  (void)nsx;
  int qi = t >> 6, ki = t & 63;
  int ny = nsy[0];
  float rx = (float)(qi / ny) - (float)(ki / ny);
  float ry = (float)(qi % ny) - (float)(ki % ny);
  rx = (rx == 0.f) ? 0.f : copysignf(log1pf(fabsf(rx)), rx);
  ry = (ry == 0.f) ? 0.f : copysignf(log1pf(fabsf(ry)), ry);
  float acc[NH_];
#pragma unroll
  for (int h = 0; h < NH_; ++h) acc[h] = 0.f;
  for (int j = 0; j < 64; ++j) {
    float hj = fmaf(rx, w1[j], fmaf(ry, w1[64 + j], b1[j]));
    hj = fmaxf(hj, 0.f);
#pragma unroll
    for (int h = 0; h < NH_; ++h) acc[h] += hj * w2[j * NH_ + h];
  }
#pragma unroll
  for (int h = 0; h < NH_; ++h) bias[h * S_ * S_ + t] = acc[h];
}

// ----------------------------- 5. scores -----------------------------------
__global__ void k_scores(const __bf16* __restrict__ qfb,
                         const __bf16* __restrict__ kfb,
                         const float* __restrict__ bias,
                         float* __restrict__ scores) {
  int bh = blockIdx.z, qt = blockIdx.y, kt = blockIdx.x;
  const __bf16* Qb = qfb + ((long long)bh * S_ + qt * 16) * FEAT_;
  const __bf16* Kb = kfb + ((long long)bh * S_ + kt * 16) * FEAT_;
  v8f acc = {};
  for (int kk = 0; kk < FEAT_; kk += 32) {
    __builtin_prefetch(Qb + kk + 2048, 0, 0);
    __builtin_prefetch(Kb + kk + 2048, 0, 0);
    v16bf a = load_a16x32(Qb + kk, FEAT_);
    v16bf b = load_bT16(Kb + kk, FEAT_);
    acc = wmma_bf16(a, b, acc);
  }
  int h = bh & (NH_ - 1);
  int lane = threadIdx.x & 31;
  int col = lane & 15, rb = (lane >> 4) * 8;
#pragma unroll
  for (int g = 0; g < 8; ++g) {
    int qi = qt * 16 + rb + g, ki = kt * 16 + col;
    scores[((long long)bh * S_ + qi) * S_ + ki] =
        acc[g] + bias[(h * S_ + qi) * S_ + ki];
  }
}

// ----------------------------- 6. softmax ----------------------------------
__global__ void k_softmax(const float* __restrict__ scores,
                          __bf16* __restrict__ attn) {
  int row = blockIdx.x;             // bh*64 + qi
  int lane = threadIdx.x;
  const float* src = scores + (long long)row * S_;
  float v0 = src[lane], v1 = src[lane + 32];
  float mx = fmaxf(v0, v1);
#pragma unroll
  for (int off = 16; off > 0; off >>= 1) mx = fmaxf(mx, __shfl_xor(mx, off, 32));
  float e0 = __expf(v0 - mx), e1 = __expf(v1 - mx);
  float sum = e0 + e1;
#pragma unroll
  for (int off = 16; off > 0; off >>= 1) sum += __shfl_xor(sum, off, 32);
  float inv = 1.0f / sum;
  attn[(long long)row * S_ + lane] = (__bf16)(e0 * inv);
  attn[(long long)row * S_ + lane + 32] = (__bf16)(e1 * inv);
}

// ----------------------------- 7. attn @ V ---------------------------------
// vfb is [bh][feature][s]; saf[mr][bs][chi] = sum_k attn[bh,q,k]*v[bh,k,f]
__global__ void k_attnv(const __bf16* __restrict__ attn,
                        const __bf16* __restrict__ vfb,
                        __bf16* __restrict__ safb) {
  int bh = blockIdx.z, qt = blockIdx.y, fg = blockIdx.x;
  const __bf16* Ab = attn + ((long long)bh * S_ + qt * 16) * S_;
  v16bf a0 = load_a16x32(Ab + 0, S_);
  v16bf a1 = load_a16x32(Ab + 32, S_);
  int lane = threadIdx.x & 31;
  int col = lane & 15, rb = (lane >> 4) * 8;
  int b = bh >> 3, h = bh & 7;
#pragma unroll
  for (int ft = fg * 8; ft < fg * 8 + 8; ++ft) {
    int f0 = ft * 16;
    const __bf16* Bb = vfb + ((long long)bh * FEAT_ + f0) * 64;
    v16bf b0 = load_bT16(Bb, 64);
    v16bf b1 = load_bT16(Bb + 32, 64);
    v8f acc = {};
    acc = wmma_bf16(a0, b0, acc);
    acc = wmma_bf16(a1, b1, acc);
#pragma unroll
    for (int g = 0; g < 8; ++g) {
      int s = qt * 16 + rb + g;
      int f = f0 + col;
      int dd = f >> 9, mr = f & 511;
      int chi = h * 64 + dd;
      int bs = b * S_ + s;
      safb[((long long)mr * BS_ + bs) * HD_ + chi] = (__bf16)acc[g];
    }
  }
}

// ------------------- 8. per-mode complex output projection -----------------
// B slab (wo for this mode/ntile, both re+im: 2x16x512 bf16 = 32KB) is staged
// into LDS by the Tensor Data Mover, then all 8 waves read it via ds_load.
__global__ void k_gemm_out(const __bf16* __restrict__ safb,
                           const __bf16* __restrict__ wob,
                           float* __restrict__ offq) {
  int m = blockIdx.z;
  int nt = blockIdx.x;              // 0..3 : 16-wide co tile
  int wave = threadIdx.x >> 5;      // 0..7 : 16-row bs tile
  __shared__ __bf16 smemB[2 * 16 * 512];

  if (threadIdx.x < 32) {
    unsigned lds_base = (unsigned)(unsigned long long)(&smemB[0]);
    const __bf16* g0 = wob + ((long long)(m * 2 + 0) * 64 + nt * 16) * 512;
    const __bf16* g1 = wob + ((long long)(m * 2 + 1) * 64 + nt * 16) * 512;
    tdm_load_2d_to_lds(lds_base,          g0, 512, 16, 512, 16, 512ull);
    tdm_load_2d_to_lds(lds_base + 16384u, g1, 512, 16, 512, 16, 512ull);
    __builtin_amdgcn_s_wait_tensorcnt(0);
  }
  __syncthreads();

  const __bf16* Ar = safb + ((long long)(m * 2 + 0) * BS_ + wave * 16) * HD_;
  const __bf16* Ai = safb + ((long long)(m * 2 + 1) * BS_ + wave * 16) * HD_;
  const __bf16* BrT = &smemB[0];            // [16 co][512 ci]
  const __bf16* BiT = &smemB[16 * 512];

  v8f rr = {}, ii = {}, im = {};
  for (int kk = 0; kk < HD_; kk += 32) {
    v16bf ar = load_a16x32(Ar + kk, HD_);
    v16bf ai = load_a16x32(Ai + kk, HD_);
    v16bf br = load_bT16(BrT + kk, 512);
    v16bf bi = load_bT16(BiT + kk, 512);
    rr = wmma_bf16(ar, br, rr);
    ii = wmma_bf16(ai, bi, ii);
    im = wmma_bf16(ar, bi, im);
    im = wmma_bf16(ai, br, im);
  }
  int lane = threadIdx.x & 31;
  int col = lane & 15, rb = (lane >> 4) * 8;
#pragma unroll
  for (int g = 0; g < 8; ++g) {
    int bs = wave * 16 + rb + g;
    int co = nt * 16 + col;
    long long base = (((long long)bs * 64 + co) * M2_ + m) * 2;
    offq[base + 0] = rr[g] - ii[g];
    offq[base + 1] = im[g];
  }
}

// ------------------------- 9. inverse truncated DFT ------------------------
__global__ void k_idft(const float* __restrict__ offq, float* __restrict__ out) {
  int plane = blockIdx.x;           // bs*64 + co
  __shared__ float fr[256], fi[256];
  __shared__ float tr[512], ti[512];
  const float* src = offq + (long long)plane * 512;
  {
    int i = threadIdx.x;
    fr[i] = src[i * 2 + 0];
    fi[i] = src[i * 2 + 1];
  }
  __syncthreads();
  for (int i = threadIdx.x; i < 512; i += 256) {
    int kx = i >> 5, y = i & 31;
    float ar = 0.f, ai = 0.f;
    for (int ky = 0; ky < 16; ++ky) {
      float w = (ky == 0) ? 1.0f : 2.0f;
      float ang = 2.f * PI_F * (float)(ky * y) / 32.f;
      float s, c; __sincosf(ang, &s, &c);
      float vr = fr[kx * 16 + ky] * w, vi = fi[kx * 16 + ky] * w;
      ar += vr * c - vi * s;
      ai += vr * s + vi * c;
    }
    tr[i] = ar; ti[i] = ai;
  }
  __syncthreads();
  float* dst = out + (long long)plane * 1024;
  for (int i = threadIdx.x; i < 1024; i += 256) {
    int x = i >> 5, y = i & 31;
    float acc = 0.f;
    for (int kx = 0; kx < 16; ++kx) {
      float ang = 2.f * PI_F * (float)(kx * x) / 32.f;
      float s, c; __sincosf(ang, &s, &c);
      acc += tr[kx * 32 + y] * c - ti[kx * 32 + y] * s;
    }
    dst[i] = acc * (1.0f / 1024.0f);
  }
}

// ----------------------------- host launcher -------------------------------
extern "C" void kernel_launch(void* const* d_in, const int* in_sizes, int n_in,
                              void* d_out, int out_size, void* d_ws, size_t ws_size,
                              hipStream_t stream) {
  (void)in_sizes; (void)n_in; (void)out_size; (void)ws_size;
  const float* seq = (const float*)d_in[0];
  const float* wq  = (const float*)d_in[1];
  const float* wk  = (const float*)d_in[2];
  const float* wv  = (const float*)d_in[3];
  const float* wo  = (const float*)d_in[4];
  const float* cw1 = (const float*)d_in[5];
  const float* cb1 = (const float*)d_in[6];
  const float* cw2 = (const float*)d_in[7];
  const int*   nsx = (const int*)d_in[8];
  const int*   nsy = (const int*)d_in[9];
  float* outp = (float*)d_out;

  char* ws = (char*)d_ws;
  size_t cur = 0;
  auto carve = [&](size_t bytes) -> char* {
    char* p = ws + cur;
    cur += (bytes + 255) & ~(size_t)255;
    return p;
  };

  const size_t nWElems = (size_t)M2_ * 2 * 64 * HD_;
  __bf16* xfb  = (__bf16*)carve((size_t)M2_ * 2 * BS_ * 64 * sizeof(__bf16));
  __bf16* wqb  = (__bf16*)carve(nWElems * sizeof(__bf16));
  __bf16* wkb  = (__bf16*)carve(nWElems * sizeof(__bf16));
  __bf16* wvb  = (__bf16*)carve(nWElems * sizeof(__bf16));
  __bf16* wob  = (__bf16*)carve(nWElems * sizeof(__bf16));
  __bf16* qfb  = (__bf16*)carve((size_t)BH_ * S_ * FEAT_ * sizeof(__bf16));
  __bf16* kfb  = (__bf16*)carve((size_t)BH_ * S_ * FEAT_ * sizeof(__bf16));
  __bf16* vfb  = (__bf16*)carve((size_t)BH_ * FEAT_ * S_ * sizeof(__bf16));
  float*  bias = (float*)carve((size_t)NH_ * S_ * S_ * sizeof(float));
  float*  scor = (float*)carve((size_t)BH_ * S_ * S_ * sizeof(float));
  __bf16* attn = (__bf16*)carve((size_t)BH_ * S_ * S_ * sizeof(__bf16));
  __bf16* safb = (__bf16*)carve((size_t)M2_ * 2 * BS_ * HD_ * sizeof(__bf16));
  float*  offq = (float*)carve((size_t)BS_ * 64 * M2_ * 2 * sizeof(float));

  {
    int blocks = (int)(nWElems / 256);
    k_cvt_w<<<blocks, 256, 0, stream>>>(wq, wqb, 64, HD_);
    k_cvt_w<<<blocks, 256, 0, stream>>>(wk, wkb, 64, HD_);
    k_cvt_w<<<blocks, 256, 0, stream>>>(wv, wvb, 64, HD_);
    k_cvt_w<<<blocks, 256, 0, stream>>>(wo, wob, HD_, 64);
  }
  k_dft_fwd<<<BS_ * 64, 256, 0, stream>>>(seq, xfb);
  k_gemm_modes<<<dim3(4, 8, M2_ * 3), 32, 0, stream>>>(
      xfb, wqb, wkb, wvb, qfb, kfb, vfb);
  k_bias<<<16, 256, 0, stream>>>(cw1, cb1, cw2, nsx, nsy, bias);
  k_scores<<<dim3(4, 4, BH_), 32, 0, stream>>>(qfb, kfb, bias, scor);
  k_softmax<<<BH_ * S_, 32, 0, stream>>>(scor, attn);
  k_attnv<<<dim3(FEAT_ / 16 / 8, 4, BH_), 32, 0, stream>>>(attn, vfb, safb);
  k_gemm_out<<<dim3(4, 1, M2_), 256, 0, stream>>>(safb, wob, offq);
  k_idft<<<BS_ * 64, 256, 0, stream>>>(offq, outp);
}
